// MipNerfModel_49924699848705
// MI455X (gfx1250) — matrix operand order repaired and checked by reading, not compiled, not run on hardware
//
#include <hip/hip_runtime.h>
#include <cstdint>
#include <cstddef>

typedef __attribute__((ext_vector_type(16))) _Float16 v16h;
typedef __attribute__((ext_vector_type(8)))  _Float16 v8h;
typedef __attribute__((ext_vector_type(8)))  float    v8f;

#define N_RAYS 1024
#define N_SAMP 128
#define M_TOTAL (N_RAYS * N_SAMP)   /* 131072 */

union AF { v16h v; v8h h[2]; };

__device__ __forceinline__ v8h load8h(const _Float16* p) { return *(const v8h*)p; }

// ---------------------------------------------------------------------------
// Generic WMMA GEMM:  out[M x N] = act( A1[M x K1] @ W1 + A2[M x K2] @ W2 + b )
// Bt is the transposed f16 weight [N][K1+K2] (row-major, ldb = K1+K2).
// Block = 8 waves = 128 rows x 128 cols; each wave owns a 16x128 strip.
// The B tile (128 cols x Ktot) is staged once per block into LDS with
// gfx1250 async global->LDS loads (ASYNCcnt), padded +8 halves per row so
// fragment reads are LDS-bank-conflict-free. A2 row = row >> a2shift
// (broadcast support); A2 may be null (K2==0).
// ---------------------------------------------------------------------------
__global__ __launch_bounds__(256) void gemm_wmma_kernel(
    const _Float16* __restrict__ A1, int lda1, int K1,
    const _Float16* __restrict__ A2, int lda2, int K2, int a2shift,
    const _Float16* __restrict__ Bt, int ldb,
    const float* __restrict__ bias,
    _Float16* __restrict__ out, int ldo, int relu)
{
  extern __shared__ _Float16 sB[];          // 128 * (ldb + 8) halves
  const int Ktot = ldb;
  const int Kpad = Ktot + 8;
  const int col0 = blockIdx.y * 128;

  // ---- stage B tile [128 x Ktot] -> LDS via async copy (16B per lane) ----
  {
    const int kchunks = Ktot >> 3;                   // 8-half (16B) chunks/row
    const int nchunk  = 128 * kchunks;
    for (int c = threadIdx.x; c < nchunk; c += 256) {
      int coll = c / kchunks;
      int koff = (c - coll * kchunks) << 3;
      const _Float16* src = Bt + (size_t)(col0 + coll) * Ktot + koff;
      unsigned lds_off = (unsigned)(uintptr_t)(sB + coll * Kpad + koff);
      unsigned long long ga = (unsigned long long)(uintptr_t)src;
      asm volatile("global_load_async_to_lds_b128 %0, %1, off"
                   :: "v"(lds_off), "v"(ga) : "memory");
    }
    asm volatile("s_wait_asynccnt 0x0" ::: "memory");
    __syncthreads();
  }

  const int lane = threadIdx.x & 31;
  const int wave = threadIdx.x >> 5;
  const int row0 = blockIdx.x * 128 + wave * 16;
  const int lidx = lane & 15;
  const int hiA  = (lane >> 4) << 3;    // 0 or 8  (A k-offset per lane group)
  const int hiB  = (lane >> 4) << 3;    // 0 or 8  (B k-offset per lane group)
  const int rowA = row0 + lidx;

  float bia[8];
  #pragma unroll
  for (int c = 0; c < 8; ++c) bia[c] = bias[col0 + c * 16 + lidx];

  v8f acc[8] = {};

  const _Float16* sBl = sB + (size_t)lidx * Kpad + hiB;  // lane's B base

  for (int kb = 0; kb < K1; kb += 32) {
    AF a;
    const _Float16* ap = A1 + (size_t)rowA * lda1 + kb + hiA;
    a.h[0] = load8h(ap);        // K = kb+hiA .. +7
    a.h[1] = load8h(ap + 16);   // K = kb+16+hiA .. +7
    #pragma unroll
    for (int c = 0; c < 8; ++c) {
      AF b;
      const _Float16* bp = sBl + (size_t)(c * 16) * Kpad + kb;
      b.h[0] = *(const v8h*)bp;
      b.h[1] = *(const v8h*)(bp + 16);
      acc[c] = __builtin_amdgcn_wmma_f32_16x16x32_f16(
          false, a.v, false, b.v, (short)0, acc[c], false, false);
    }
  }
  if (K2 > 0) {
    const int rowA2 = rowA >> a2shift;
    for (int kb = 0; kb < K2; kb += 32) {
      AF a;
      const _Float16* ap = A2 + (size_t)rowA2 * lda2 + kb + hiA;
      a.h[0] = load8h(ap);
      a.h[1] = load8h(ap + 16);
      #pragma unroll
      for (int c = 0; c < 8; ++c) {
        AF b;
        const _Float16* bp = sBl + (size_t)(c * 16) * Kpad + K1 + kb;
        b.h[0] = *(const v8h*)bp;
        b.h[1] = *(const v8h*)(bp + 16);
        acc[c] = __builtin_amdgcn_wmma_f32_16x16x32_f16(
            false, a.v, false, b.v, (short)0, acc[c], false, false);
      }
    }
  }
  const int rbase = row0 + ((lane >> 4) << 3);   // D: lanes 0-15 -> M 0..7, 16-31 -> M 8..15
  #pragma unroll
  for (int c = 0; c < 8; ++c) {
    const int colD = col0 + c * 16 + lidx;
    #pragma unroll
    for (int r = 0; r < 8; ++r) {
      float v = acc[c][r] + bia[c];
      if (relu) v = fmaxf(v, 0.0f);
      out[(size_t)(rbase + r) * ldo + colD] = (_Float16)v;
    }
  }
}

// ---------------------------------------------------------------------------
// Weight convert: fp32 W[K x N] -> f16 Bt[N x Kpad] (transposed, zero-padded K)
// ---------------------------------------------------------------------------
__global__ __launch_bounds__(256) void convert_wt(const float* __restrict__ W,
    _Float16* __restrict__ Bt, int K, int N, int Kpad)
{
  int tid = blockIdx.x * 256 + threadIdx.x;
  int total = N * Kpad;
  if (tid >= total) return;
  int n = tid / Kpad, k = tid % Kpad;
  Bt[tid] = (k < K) ? (_Float16)W[(size_t)k * N + n] : (_Float16)0.0f;
}

// ---------------------------------------------------------------------------
// t_vals init (level 0): near + (far-near) * j/128
// ---------------------------------------------------------------------------
__global__ __launch_bounds__(256) void tv_init(const float* __restrict__ nearp,
    const float* __restrict__ farp, float* __restrict__ tv)
{
  int tid = blockIdx.x * 256 + threadIdx.x;
  if (tid >= N_RAYS * 129) return;
  int i = tid / 129, j = tid % 129;
  tv[tid] = nearp[i] + (farp[i] - nearp[i]) * ((float)j / 128.0f);
}

// ---------------------------------------------------------------------------
// View-dir positional encoding -> f16 [N_RAYS x 32] (27 valid cols + zero pad)
// ---------------------------------------------------------------------------
__global__ __launch_bounds__(256) void vdir_enc_kernel(const float* __restrict__ vd,
    _Float16* __restrict__ vdenc)
{
  int i = blockIdx.x * 256 + threadIdx.x;
  if (i >= N_RAYS) return;
  float v[3] = { vd[i * 3], vd[i * 3 + 1], vd[i * 3 + 2] };
  _Float16 e[32];
  #pragma unroll
  for (int k = 0; k < 3; ++k) e[k] = (_Float16)v[k];
  float scale = 1.0f;
  #pragma unroll
  for (int d = 0; d < 4; ++d) {
    #pragma unroll
    for (int k = 0; k < 3; ++k) {
      float y = v[k] * scale;
      e[3 + d * 3 + k]      = (_Float16)__sinf(y);
      e[3 + 12 + d * 3 + k] = (_Float16)__cosf(y);
    }
    scale *= 2.0f;
  }
  #pragma unroll
  for (int j = 27; j < 32; ++j) e[j] = (_Float16)0.0f;
  v8h* dst = (v8h*)(vdenc + (size_t)i * 32);
  const v8h* src = (const v8h*)e;
  #pragma unroll
  for (int j = 0; j < 4; ++j) dst[j] = src[j];
}

// ---------------------------------------------------------------------------
// Conical frustum cast + integrated positional encoding -> f16 [M x 96]
// ---------------------------------------------------------------------------
__global__ __launch_bounds__(256) void cast_ipe(const float* __restrict__ origins,
    const float* __restrict__ dirs, const float* __restrict__ radii,
    const float* __restrict__ tv, _Float16* __restrict__ enc)
{
  int idx = blockIdx.x * 256 + threadIdx.x;
  if (idx >= M_TOTAL) return;
  int ray = idx >> 7, s = idx & 127;
  float t0 = tv[ray * 129 + s], t1 = tv[ray * 129 + s + 1];
  float mu = 0.5f * (t0 + t1), hw = 0.5f * (t1 - t0);
  float mu2 = mu * mu, hw2 = hw * hw, hw4 = hw2 * hw2;
  float denom = 3.0f * mu2 + hw2;
  float t_mean = mu + 2.0f * mu * hw2 / denom;
  float t_var  = hw2 / 3.0f - (4.0f / 15.0f) * (hw4 * (12.0f * mu2 - hw2)) / (denom * denom);
  float rr = radii[ray];
  float r_var = rr * rr * (mu2 * 0.25f + (5.0f / 12.0f) * hw2 - (4.0f / 15.0f) * hw4 / denom);
  float d[3] = { dirs[ray * 3], dirs[ray * 3 + 1], dirs[ray * 3 + 2] };
  float dmag = fmaxf(1e-10f, d[0] * d[0] + d[1] * d[1] + d[2] * d[2]);
  float mean[3], cov[3];
  #pragma unroll
  for (int k = 0; k < 3; ++k) {
    mean[k] = d[k] * t_mean + origins[ray * 3 + k];
    float od = d[k] * d[k];
    cov[k] = t_var * od + r_var * (1.0f - od / dmag);
  }
  _Float16 e[96];
  float scale = 1.0f;
  #pragma unroll
  for (int dg = 0; dg < 16; ++dg) {
    float s2 = scale * scale;
    #pragma unroll
    for (int k = 0; k < 3; ++k) {
      float y   = mean[k] * scale;
      float att = __expf(-0.5f * cov[k] * s2);
      e[dg * 3 + k]      = (_Float16)(att * __sinf(y));
      e[48 + dg * 3 + k] = (_Float16)(att * __cosf(y));
    }
    scale *= 2.0f;
  }
  v8h* dst = (v8h*)(enc + (size_t)idx * 96);
  const v8h* src = (const v8h*)e;
  #pragma unroll
  for (int j = 0; j < 12; ++j) dst[j] = src[j];
}

// ---------------------------------------------------------------------------
// Density head: x[256] . wd + bd -> softplus(raw - 1)
// ---------------------------------------------------------------------------
__global__ __launch_bounds__(256) void density_head(const _Float16* __restrict__ X,
    const float* __restrict__ wdp, const float* __restrict__ bdp,
    float* __restrict__ density)
{
  int r = blockIdx.x * 256 + threadIdx.x;
  if (r >= M_TOTAL) return;
  const _Float16* x = X + (size_t)r * 256;
  float s = bdp[0];
  for (int k = 0; k < 256; ++k) s += (float)x[k] * wdp[k];
  s += -1.0f;  // DENSITY_BIAS
  density[r] = (s > 20.0f) ? s : log1pf(__expf(s));
}

// ---------------------------------------------------------------------------
// RGB head: h[128] @ wr[128x3] + br -> sigmoid * 1.002 - 0.001
// ---------------------------------------------------------------------------
__global__ __launch_bounds__(256) void rgb_head(const _Float16* __restrict__ H,
    const float* __restrict__ wrp, const float* __restrict__ brp,
    float* __restrict__ rgb)
{
  int r = blockIdx.x * 256 + threadIdx.x;
  if (r >= M_TOTAL) return;
  const _Float16* h = H + (size_t)r * 128;
  float s0 = brp[0], s1 = brp[1], s2 = brp[2];
  for (int k = 0; k < 128; ++k) {
    float hv = (float)h[k];
    s0 += hv * wrp[k * 3 + 0];
    s1 += hv * wrp[k * 3 + 1];
    s2 += hv * wrp[k * 3 + 2];
  }
  float o[3] = { s0, s1, s2 };
  #pragma unroll
  for (int c = 0; c < 3; ++c) {
    float v = 1.0f / (1.0f + __expf(-o[c]));
    rgb[(size_t)r * 3 + c] = v * 1.002f - 0.001f;
  }
}

// ---------------------------------------------------------------------------
// Volumetric rendering: one thread per ray (serial transmittance scan)
// ---------------------------------------------------------------------------
__global__ __launch_bounds__(256) void volrender(const float* __restrict__ tv,
    const float* __restrict__ density, const float* __restrict__ rgb,
    const float* __restrict__ dirs, float* __restrict__ wout,
    float* __restrict__ out)
{
  int i = blockIdx.x * 256 + threadIdx.x;
  if (i >= N_RAYS) return;
  const float* tvr = tv + i * 129;
  float d0 = dirs[i * 3], d1 = dirs[i * 3 + 1], d2 = dirs[i * 3 + 2];
  float nrm = sqrtf(d0 * d0 + d1 * d1 + d2 * d2);
  float T = 1.0f, acc = 0.0f, c0 = 0.0f, c1 = 0.0f, c2 = 0.0f, wt = 0.0f;
  for (int s = 0; s < 128; ++s) {
    float ta = tvr[s], tb = tvr[s + 1];
    float dd = density[i * 128 + s] * ((tb - ta) * nrm);
    float e  = __expf(-dd);
    float w  = (1.0f - e) * T;
    T *= e;
    wout[i * 128 + s] = w;
    const float* rg = rgb + (size_t)(i * 128 + s) * 3;
    c0 += w * rg[0]; c1 += w * rg[1]; c2 += w * rg[2];
    acc += w;
    wt += w * 0.5f * (ta + tb);
  }
  float dist = wt / acc;
  if (!(dist == dist)) dist = __builtin_inff();
  dist = fminf(fmaxf(dist, tvr[0]), tvr[128]);
  c0 += 1.0f - acc; c1 += 1.0f - acc; c2 += 1.0f - acc;  // white background
  out[i * 3 + 0] = c0;
  out[i * 3 + 1] = c1;
  out[i * 3 + 2] = c2;
  out[3072 + i] = dist;
  out[4096 + i] = acc;
}

// ---------------------------------------------------------------------------
// Inverse-CDF resampling (level 1 t_vals): one thread per ray, merge scan
// ---------------------------------------------------------------------------
__global__ __launch_bounds__(256) void resample_kernel(const float* __restrict__ tvin,
    const float* __restrict__ wts, float* __restrict__ tvout)
{
  int i = blockIdx.x * 256 + threadIdx.x;
  if (i >= N_RAYS) return;
  const float* w = wts + i * 128;
  float blur[128];
  float wsum = 0.0f;
  for (int j = 0; j < 128; ++j) {
    int a = (j - 1 < 0) ? 0 : j - 1;
    int c = (j + 1 > 127) ? 127 : j + 1;
    float m0 = fmaxf(w[a], w[j]);
    float m1 = fmaxf(w[j], w[c]);
    blur[j] = 0.5f * (m0 + m1) + 0.01f;   // RESAMPLE_PADDING
    wsum += blur[j];
  }
  float padv = fmaxf(0.0f, 1e-5f - wsum);
  float add = padv / 128.0f;
  wsum += padv;
  float cdf[129];
  cdf[0] = 0.0f;
  float run = 0.0f;
  for (int j = 0; j < 127; ++j) {
    run += (blur[j] + add) / wsum;
    cdf[j + 1] = fminf(1.0f, run);
  }
  cdf[128] = 1.0f;
  const float* bins = tvin + i * 129;
  float* o = tvout + i * 129;
  const float step = (1.0f - 1.1920929e-7f) / 128.0f;
  int lo = 0;
  for (int j = 0; j < 129; ++j) {
    float u = (float)j * step;
    while (lo < 127 && cdf[lo + 1] <= u) ++lo;
    float cl = cdf[lo], ch = cdf[lo + 1];
    float t = (u - cl) / (ch - cl);
    if (!(t == t)) t = 0.0f;
    t = fminf(fmaxf(t, 0.0f), 1.0f);
    o[j] = bins[lo] + t * (bins[lo + 1] - bins[lo]);
  }
}

// ---------------------------------------------------------------------------
extern "C" void kernel_launch(void* const* d_in, const int* in_sizes, int n_in,
                              void* d_out, int out_size, void* d_ws, size_t ws_size,
                              hipStream_t stream)
{
  (void)in_sizes; (void)n_in; (void)out_size; (void)ws_size;
  const float* origins    = (const float*)d_in[0];
  const float* directions = (const float*)d_in[1];
  const float* viewdirs   = (const float*)d_in[2];
  const float* radii      = (const float*)d_in[3];
  const float* nearp      = (const float*)d_in[4];
  const float* farp       = (const float*)d_in[5];
  const float* W[8]; const float* Bv[8];
  for (int i = 0; i < 8; ++i) { W[i] = (const float*)d_in[6 + 2 * i]; Bv[i] = (const float*)d_in[7 + 2 * i]; }
  const float* wd = (const float*)d_in[22]; const float* bd = (const float*)d_in[23];
  const float* wb = (const float*)d_in[24]; const float* bb = (const float*)d_in[25];
  const float* wc = (const float*)d_in[26]; const float* bc = (const float*)d_in[27];
  const float* wr = (const float*)d_in[28]; const float* br = (const float*)d_in[29];

  char* base = (char*)d_ws;
  size_t off = 0;
  auto take = [&](size_t bytes) -> char* {
    char* p = base + off;
    off = (off + bytes + 255) & ~(size_t)255;
    return p;
  };
  const int M = M_TOTAL;
  _Float16* enc   = (_Float16*)take((size_t)M * 96 * 2);
  _Float16* Xb    = (_Float16*)take((size_t)M * 256 * 2);
  _Float16* Yb    = (_Float16*)take((size_t)M * 256 * 2);
  _Float16* Hb    = (_Float16*)take((size_t)M * 128 * 2);
  float* density  = (float*)take((size_t)M * 4);
  float* rgb      = (float*)take((size_t)M * 3 * 4);
  float* wbuf     = (float*)take((size_t)M * 4);
  float* tv0      = (float*)take((size_t)N_RAYS * 129 * 4);
  float* tv1      = (float*)take((size_t)N_RAYS * 129 * 4);
  _Float16* vdenc = (_Float16*)take((size_t)N_RAYS * 32 * 2);
  const int Kdims[8] = { 96, 256, 256, 256, 256, 352, 256, 256 };
  _Float16* wt[8];
  for (int i = 0; i < 8; ++i) wt[i] = (_Float16*)take((size_t)256 * Kdims[i] * 2);
  _Float16* wbt = (_Float16*)take((size_t)256 * 256 * 2);
  _Float16* wct = (_Float16*)take((size_t)128 * 288 * 2);

  // --- weight staging: fp32 [K x N] -> f16 transposed [N x Kpad] ---
  for (int i = 0; i < 8; ++i) {
    int total = 256 * Kdims[i];
    convert_wt<<<(total + 255) / 256, 256, 0, stream>>>(W[i], wt[i], Kdims[i], 256, Kdims[i]);
  }
  convert_wt<<<(256 * 256 + 255) / 256, 256, 0, stream>>>(wb, wbt, 256, 256, 256);
  convert_wt<<<(128 * 288 + 255) / 256, 256, 0, stream>>>(wc, wct, 283, 128, 288);

  tv_init<<<(N_RAYS * 129 + 255) / 256, 256, 0, stream>>>(nearp, farp, tv0);
  vdir_enc_kernel<<<(N_RAYS + 255) / 256, 256, 0, stream>>>(viewdirs, vdenc);

  auto gemm = [&](const _Float16* A1, int lda1, int K1,
                  const _Float16* A2, int lda2, int K2, int shift,
                  const _Float16* Bt, int ldb, const float* bias,
                  _Float16* o, int ldo, int N, int relu) {
    dim3 g(M / 128, N / 128);
    size_t smem = (size_t)128 * (ldb + 8) * 2;   // padded B tile in LDS
    gemm_wmma_kernel<<<g, 256, smem, stream>>>(A1, lda1, K1, A2, lda2, K2, shift,
                                               Bt, ldb, bias, o, ldo, relu);
  };

  for (int lvl = 0; lvl < 2; ++lvl) {
    if (lvl == 1)
      resample_kernel<<<(N_RAYS + 255) / 256, 256, 0, stream>>>(tv0, wbuf, tv1);
    const float* tv = (lvl == 0) ? tv0 : tv1;
    cast_ipe<<<M / 256, 256, 0, stream>>>(origins, directions, radii, tv, enc);

    // trunk: enc->L0->X->L1->Y->L2->X->L3->Y->L4->X; concat(X,enc)->L5->Y->L6->X->L7->Y
    gemm(enc, 96, 96,   nullptr, 0, 0, 0, wt[0], 96,  Bv[0], Xb, 256, 256, 1);
    gemm(Xb, 256, 256,  nullptr, 0, 0, 0, wt[1], 256, Bv[1], Yb, 256, 256, 1);
    gemm(Yb, 256, 256,  nullptr, 0, 0, 0, wt[2], 256, Bv[2], Xb, 256, 256, 1);
    gemm(Xb, 256, 256,  nullptr, 0, 0, 0, wt[3], 256, Bv[3], Yb, 256, 256, 1);
    gemm(Yb, 256, 256,  nullptr, 0, 0, 0, wt[4], 256, Bv[4], Xb, 256, 256, 1);
    gemm(Xb, 256, 256,  enc, 96, 96, 0,   wt[5], 352, Bv[5], Yb, 256, 256, 1);
    gemm(Yb, 256, 256,  nullptr, 0, 0, 0, wt[6], 256, Bv[6], Xb, 256, 256, 1);
    gemm(Xb, 256, 256,  nullptr, 0, 0, 0, wt[7], 256, Bv[7], Yb, 256, 256, 1);

    density_head<<<M / 256, 256, 0, stream>>>(Yb, wd, bd, density);
    gemm(Yb, 256, 256,  nullptr, 0, 0, 0, wbt, 256, bb, Xb, 256, 256, 0);        // bottleneck (no relu)
    gemm(Xb, 256, 256,  vdenc, 32, 32, 7, wct, 288, bc, Hb, 128, 128, 1);        // color hidden
    rgb_head<<<M / 256, 256, 0, stream>>>(Hb, wr, br, rgb);

    volrender<<<(N_RAYS + 255) / 256, 256, 0, stream>>>(tv, density, rgb, directions,
                                                        wbuf, (float*)d_out + lvl * 5120);
  }
}
